// Phyla_57423712747978
// MI455X (gfx1250) — matrix operand order repaired and checked by reading, not compiled, not run on hardware
//
#include <hip/hip_runtime.h>
#include <hip/hip_bf16.h>
#include <math.h>

#define D_MODEL 256
#define N_LAYER 16
#define D_INNER 512
#define D_STATE 16
#define D_CONV  4
#define DT_RANK 16
#define BATCH   2
#define SEQ     2048
#define BL      (BATCH * SEQ)

typedef __attribute__((ext_vector_type(16))) _Float16 v16h;
typedef __attribute__((ext_vector_type(8)))  _Float16 v8h_t;
typedef __attribute__((ext_vector_type(8)))  float    v8f;

static __device__ __forceinline__ float sigmoidf_(float v) {
    return 1.0f / (1.0f + __expf(-v));
}

static __device__ __forceinline__ v16h cat8(v8h_t lo, v8h_t hi) {
    return __builtin_shufflevector(lo, hi, 0, 1, 2, 3, 4, 5, 6, 7,
                                           8, 9, 10, 11, 12, 13, 14, 15);
}

// ---------------------------------------------------------------------------
// WMMA GEMM: C[M x N] = A[M x K] (f16, row-major, lda) x W[N x K]^T (f16, ldw)
// One wave32 computes a 16x32 slab (two 16x16 N-tiles sharing the A fragment,
// two v_wmma per K-step). K % 32 == 0, M % 16 == 0; N ragged.
// Out-of-range B columns are handled by CLAMPING the row index (keeps loads
// unconditional and b128-vectorizable); garbage columns only affect outputs
// that the store mask drops, so no zero-fill is needed.
// epi: 0 = plain store; 1 = softplus(v + bias[n]) (dt_proj fusion).
// ---------------------------------------------------------------------------
__global__ __launch_bounds__(256)
void wmma_gemm_f16(const _Float16* __restrict__ A, int lda,
                   const _Float16* __restrict__ W, int ldw,
                   float* __restrict__ Cout, int ldc,
                   int M, int N, int K,
                   const float* __restrict__ bias, int epi)
{
    const int lane    = threadIdx.x & 31;
    const int wave    = threadIdx.x >> 5;
    const int tilesN  = (N + 15) >> 4;
    const int tilesN2 = (tilesN + 1) >> 1;          // 2 N-tiles per wave
    const int tilesM  = M >> 4;
    const int tile = blockIdx.x * (blockDim.x >> 5) + wave;   // wave-uniform
    if (tile >= tilesM * tilesN2) return;                     // whole-wave exit
    const int tm = tile / tilesN2;
    const int tn = (tile - tm * tilesN2) << 1;
    const int mBase  = tm << 4;
    const int nBase0 = tn << 4;
    const int nBase1 = nBase0 + 16;

    // A layout (16-bit A 16x32, ISA 7.12.2): lanes 0-15 -> K 0..7 & 16..23,
    // lanes 16-31 -> K 8..15 & 24..31; row M = lane&15. Two 16B runs per lane.
    const int mRow  = lane & 15;
    const int kGrpA = (lane >> 4) << 3;   // 0 or 8
    // B layout (32x16): lane&15 = column N; lanes 0-15 hold K 0..7 & 16..23,
    // lanes 16-31 hold K 8..15 & 24..31 (matches A). Two 16B runs per lane.
    const int nCol  = lane & 15;
    const int kGrpB = (lane >> 4) << 3;   // 0 or 8

    const int n0 = nBase0 + nCol;
    const int n1 = nBase1 + nCol;
    const int n0c = (n0 < N) ? n0 : (N - 1);
    const int n1c = (n1 < N) ? n1 : (N - 1);

    const _Float16* Abase = A + (size_t)(mBase + mRow) * lda;
    const _Float16* W0    = W + (size_t)n0c * ldw;
    const _Float16* W1    = W + (size_t)n1c * ldw;

    v8f acc0 = {}, acc1 = {};
    for (int k0 = 0; k0 < K; k0 += 32) {
        // prefetch next K-step (speculative; safe past the end)
        __builtin_prefetch(Abase + k0 + 32, 0, 0);
        __builtin_prefetch(W0 + k0 + 32, 0, 0);
        __builtin_prefetch(W1 + k0 + 32, 0, 0);

        v16h a  = cat8(*(const v8h_t*)(Abase + k0 + kGrpA),
                       *(const v8h_t*)(Abase + k0 + 16 + kGrpA));
        v16h b0 = cat8(*(const v8h_t*)(W0 + k0 + kGrpB),
                       *(const v8h_t*)(W0 + k0 + kGrpB + 16));
        v16h b1 = cat8(*(const v8h_t*)(W1 + k0 + kGrpB),
                       *(const v8h_t*)(W1 + k0 + kGrpB + 16));
        // (neg_a, A, neg_b, B, c_mod, C, reuse_a, reuse_b)
        acc0 = __builtin_amdgcn_wmma_f32_16x16x32_f16(false, a, false, b0,
                                                      (short)0, acc0, false, false);
        acc1 = __builtin_amdgcn_wmma_f32_16x16x32_f16(false, a, false, b1,
                                                      (short)0, acc1, false, false);
    }

    // C/D layout: VGPR r, lanes 0-15 -> M=r, lanes 16-31 -> M=8+r; N = lane&15.
    const int mOff = (lane >> 4) << 3;
    if (n0 < N) {
        float bv = (epi == 1) ? bias[n0] : 0.0f;
#pragma unroll
        for (int r = 0; r < 8; ++r) {
            float v = acc0[r];
            if (epi == 1) { v += bv; v = (v > 20.0f) ? v : log1pf(__expf(v)); }
            Cout[(size_t)(mBase + mOff + r) * ldc + n0] = v;
        }
    }
    if (n1 < N) {
        float bv = (epi == 1) ? bias[n1] : 0.0f;
#pragma unroll
        for (int r = 0; r < 8; ++r) {
            float v = acc1[r];
            if (epi == 1) { v += bv; v = (v > 20.0f) ? v : log1pf(__expf(v)); }
            Cout[(size_t)(mBase + mOff + r) * ldc + n1] = v;
        }
    }
}

// ---------------------------------------------------------------------------
// Elementwise / conversion helpers
// ---------------------------------------------------------------------------
__global__ void f32_to_f16_kernel(const float* __restrict__ src,
                                  _Float16* __restrict__ dst, int n)
{
    int i = blockIdx.x * blockDim.x + threadIdx.x;
    if (i < n) dst[i] = (_Float16)src[i];
}

// dt_proj_w (rows x 16) -> (rows x 32) zero-padded f16
__global__ void pad_k16to32_kernel(const float* __restrict__ src,
                                   _Float16* __restrict__ dst, int rows)
{
    int i = blockIdx.x * blockDim.x + threadIdx.x;
    if (i >= rows * 32) return;
    int r = i >> 5, k = i & 31;
    dst[i] = (k < DT_RANK) ? (_Float16)src[r * DT_RANK + k] : (_Float16)0;
}

// dbc (BL x 48) first 16 cols -> dtr16 (BL x 32), zero-padded
__global__ void dtr_pack_kernel(const float* __restrict__ dbc,
                                _Float16* __restrict__ dtr, int rows)
{
    int i = blockIdx.x * blockDim.x + threadIdx.x;
    if (i >= rows * 32) return;
    int r = i >> 5, k = i & 31;
    dtr[i] = (k < DT_RANK) ? (_Float16)dbc[r * 48 + k] : (_Float16)0;
}

__global__ void embed_kernel(const int* __restrict__ ids,
                             const float* __restrict__ emb,
                             float* __restrict__ h)
{
    int i = blockIdx.x * blockDim.x + threadIdx.x;
    if (i >= BL * D_MODEL) return;
    int tok = i >> 8;            // D_MODEL == 256
    int c   = i & (D_MODEL - 1);
    h[i] = emb[(size_t)ids[tok] * D_MODEL + c];
}

// residual = first ? h : residual + h ;  out16 = rmsnorm(residual) * w   (f16)
__global__ __launch_bounds__(256)
void resid_rmsnorm_kernel(const float* __restrict__ h,
                          float* __restrict__ resid,
                          const float* __restrict__ w,
                          _Float16* __restrict__ out16,
                          int rows, int first)
{
    int lane = threadIdx.x & 31;
    int row  = blockIdx.x * (blockDim.x >> 5) + (threadIdx.x >> 5);
    if (row >= rows) return;
    const float* hr = h + (size_t)row * D_MODEL;
    float*       rr = resid + (size_t)row * D_MODEL;
    float vals[8];
    float ss = 0.0f;
#pragma unroll
    for (int j = 0; j < 8; ++j) {
        int c = lane + j * 32;
        float v = hr[c] + (first ? 0.0f : rr[c]);
        vals[j] = v;
        rr[c] = v;
        ss += v * v;
    }
#pragma unroll
    for (int off = 16; off; off >>= 1) ss += __shfl_xor(ss, off, 32);
    float scale = rsqrtf(ss * (1.0f / (float)D_MODEL) + 1e-5f);
    _Float16* o = out16 + (size_t)row * D_MODEL;
#pragma unroll
    for (int j = 0; j < 8; ++j) {
        int c = lane + j * 32;
        o[c] = (_Float16)(vals[j] * scale * w[c]);
    }
}

// depthwise causal conv over L (window 4) + SiLU; reads x-half of xz
__global__ void conv_silu_kernel(const float* __restrict__ xz,
                                 const float* __restrict__ cw,
                                 const float* __restrict__ cb,
                                 float* __restrict__ x,
                                 _Float16* __restrict__ x16)
{
    int i = blockIdx.x * blockDim.x + threadIdx.x;
    if (i >= BL * D_INNER) return;
    int d  = i & (D_INNER - 1);
    int bt = i >> 9;                   // b*SEQ + t
    int t  = bt & (SEQ - 1);
    float acc = cb[d];
#pragma unroll
    for (int j = 0; j < D_CONV; ++j) {
        int tt = t - (D_CONV - 1) + j;
        if (tt >= 0)
            acc += cw[d * D_CONV + j] *
                   xz[(size_t)(bt - (D_CONV - 1) + j) * (2 * D_INNER) + d];
    }
    float s = acc * sigmoidf_(acc);
    x[i]   = s;
    x16[i] = (_Float16)s;
}

// Selective scan: 16 lanes per (b, d) channel, one per d_state.
// h_t = exp(dt*A) * h_{t-1} + dt*B_s*x ; y = sum_s h_t * C_s (shfl-16 reduce)
// then y = (y + D*x) * silu(z). dA/dBx computed on the fly (never materialized).
__global__ __launch_bounds__(256)
void scan_kernel(const float* __restrict__ dt,
                 const float* __restrict__ x,
                 const float* __restrict__ dbc,
                 const float* __restrict__ xz,
                 const float* __restrict__ a_log,
                 const float* __restrict__ Dskip,
                 float* __restrict__ y,
                 _Float16* __restrict__ y16)
{
    int grp = blockIdx.x * (blockDim.x >> 4) + (threadIdx.x >> 4); // b*D_INNER + d
    int s   = threadIdx.x & 15;
    if (grp >= BATCH * D_INNER) return;
    int b = grp >> 9;                  // D_INNER == 512
    int d = grp & (D_INNER - 1);
    const float Acoef = -__expf(a_log[(size_t)d * D_STATE + s]);
    const float Dv = Dskip[d];
    float hstate = 0.0f;
    const size_t baseTok = (size_t)b * SEQ;
    for (int t = 0; t < SEQ; ++t) {
        const size_t tok = baseTok + t;
        float dtv = dt[tok * D_INNER + d];
        float xv  = x[tok * D_INNER + d];
        float Bs  = dbc[tok * 48 + DT_RANK + s];
        float Cs  = dbc[tok * 48 + DT_RANK + D_STATE + s];
        hstate = __expf(dtv * Acoef) * hstate + dtv * Bs * xv;
        float contrib = hstate * Cs;
#pragma unroll
        for (int off = 8; off; off >>= 1) contrib += __shfl_xor(contrib, off, 16);
        if (s == 0) {
            float zv = xz[tok * (2 * D_INNER) + D_INNER + d];
            float yv = (contrib + Dv * xv) * (zv * sigmoidf_(zv));
            y[tok * D_INNER + d]   = yv;
            y16[tok * D_INNER + d] = (_Float16)yv;
        }
    }
}

// ---------------------------------------------------------------------------
// Host orchestration
// ---------------------------------------------------------------------------
extern "C" void kernel_launch(void* const* d_in, const int* in_sizes, int n_in,
                              void* d_out, int out_size, void* d_ws, size_t ws_size,
                              hipStream_t stream)
{
    (void)in_sizes; (void)n_in; (void)out_size; (void)ws_size;

    const int*   ids    = (const int*)  d_in[0];
    const float* emb    = (const float*)d_in[1];
    const float* in_w   = (const float*)d_in[2];
    const float* conv_w = (const float*)d_in[3];
    const float* conv_b = (const float*)d_in[4];
    const float* x_w    = (const float*)d_in[5];
    const float* dt_w   = (const float*)d_in[6];
    const float* dt_b   = (const float*)d_in[7];
    const float* a_log  = (const float*)d_in[8];
    const float* Dp     = (const float*)d_in[9];
    const float* out_w  = (const float*)d_in[10];
    const float* norm_w = (const float*)d_in[11];
    const float* norm_f = (const float*)d_in[12];
    const float* lm_w   = (const float*)d_in[13];
    float* out = (float*)d_out;

    // bump allocator over workspace
    char* cur = (char*)d_ws;
    auto alloc = [&](size_t nbytes) -> void* {
        void* p = (void*)cur;
        cur += (nbytes + 255) & ~(size_t)255;
        return p;
    };

    _Float16* in_w16 = (_Float16*)alloc((size_t)N_LAYER * 2 * D_INNER * D_MODEL * 2);
    _Float16* xw16   = (_Float16*)alloc((size_t)N_LAYER * 48 * D_INNER * 2);
    _Float16* dtw16  = (_Float16*)alloc((size_t)N_LAYER * D_INNER * 32 * 2);
    _Float16* ow16   = (_Float16*)alloc((size_t)N_LAYER * D_MODEL * D_INNER * 2);
    _Float16* lm16   = (_Float16*)alloc((size_t)20 * D_MODEL * 2);
    float*    h      = (float*)   alloc((size_t)BL * D_MODEL * 4);
    float*    resid  = (float*)   alloc((size_t)BL * D_MODEL * 4);
    _Float16* hn16   = (_Float16*)alloc((size_t)BL * D_MODEL * 2);
    float*    xz     = (float*)   alloc((size_t)BL * 2 * D_INNER * 4);
    float*    xbuf   = (float*)   alloc((size_t)BL * D_INNER * 4);
    _Float16* x16    = (_Float16*)alloc((size_t)BL * D_INNER * 2);
    float*    dbc    = (float*)   alloc((size_t)BL * 48 * 4);
    _Float16* dtr16  = (_Float16*)alloc((size_t)BL * 32 * 2);
    float*    dtbuf  = (float*)   alloc((size_t)BL * D_INNER * 4);
    float*    ybuf   = (float*)   alloc((size_t)BL * D_INNER * 4);
    _Float16* y16    = (_Float16*)alloc((size_t)BL * D_INNER * 2);

    const int TPB = 256;
    auto cdiv = [](int a, int b) { return (a + b - 1) / b; };

    auto launch_gemm = [&](const _Float16* Ap, int lda, const _Float16* Wp, int ldw,
                           float* Cp, int ldc, int M, int N, int K,
                           const float* biasp, int epi) {
        int tilesN2 = ((N + 15) / 16 + 1) / 2;
        int tiles   = (M / 16) * tilesN2;
        wmma_gemm_f16<<<cdiv(tiles, 8), TPB, 0, stream>>>(
            Ap, lda, Wp, ldw, Cp, ldc, M, N, K, biasp, epi);
    };

    // weight conversions (every launch; deterministic, graph-safe)
    {
        int n = N_LAYER * 2 * D_INNER * D_MODEL;
        f32_to_f16_kernel<<<cdiv(n, TPB), TPB, 0, stream>>>(in_w, in_w16, n);
        n = N_LAYER * 48 * D_INNER;
        f32_to_f16_kernel<<<cdiv(n, TPB), TPB, 0, stream>>>(x_w, xw16, n);
        n = N_LAYER * D_MODEL * D_INNER;
        f32_to_f16_kernel<<<cdiv(n, TPB), TPB, 0, stream>>>(out_w, ow16, n);
        n = 20 * D_MODEL;
        f32_to_f16_kernel<<<cdiv(n, TPB), TPB, 0, stream>>>(lm_w, lm16, n);
        int rows = N_LAYER * D_INNER;
        pad_k16to32_kernel<<<cdiv(rows * 32, TPB), TPB, 0, stream>>>(dt_w, dtw16, rows);
    }

    embed_kernel<<<cdiv(BL * D_MODEL, TPB), TPB, 0, stream>>>(ids, emb, h);

    for (int i = 0; i < N_LAYER; ++i) {
        resid_rmsnorm_kernel<<<BL / 8, TPB, 0, stream>>>(
            h, resid, norm_w + (size_t)i * D_MODEL, hn16, BL, (i == 0) ? 1 : 0);

        // in_proj: (BL x 256) x (256 x 1024)
        launch_gemm(hn16, D_MODEL, in_w16 + (size_t)i * 2 * D_INNER * D_MODEL, D_MODEL,
                    xz, 2 * D_INNER, BL, 2 * D_INNER, D_MODEL, nullptr, 0);

        conv_silu_kernel<<<cdiv(BL * D_INNER, TPB), TPB, 0, stream>>>(
            xz, conv_w + (size_t)i * D_INNER * D_CONV, conv_b + (size_t)i * D_INNER,
            xbuf, x16);

        // x_proj: (BL x 512) x (512 x 48)
        launch_gemm(x16, D_INNER, xw16 + (size_t)i * 48 * D_INNER, D_INNER,
                    dbc, 48, BL, 48, D_INNER, nullptr, 0);

        dtr_pack_kernel<<<cdiv(BL * 32, TPB), TPB, 0, stream>>>(dbc, dtr16, BL);

        // dt_proj + bias + softplus fused epilogue: (BL x 32) x (32 x 512)
        launch_gemm(dtr16, 32, dtw16 + (size_t)i * D_INNER * 32, 32,
                    dtbuf, D_INNER, BL, D_INNER, 32, dt_b + (size_t)i * D_INNER, 1);

        scan_kernel<<<cdiv(BATCH * D_INNER * 16, TPB), TPB, 0, stream>>>(
            dtbuf, xbuf, dbc, xz,
            a_log + (size_t)i * D_INNER * D_STATE, Dp + (size_t)i * D_INNER,
            ybuf, y16);

        // out_proj: (BL x 512) x (512 x 256) -> new h
        launch_gemm(y16, D_INNER, ow16 + (size_t)i * D_MODEL * D_INNER, D_INNER,
                    h, D_MODEL, BL, D_MODEL, D_INNER, nullptr, 0);
    }

    // final residual + rmsnorm(norm_f) -> hn16
    resid_rmsnorm_kernel<<<BL / 8, TPB, 0, stream>>>(h, resid, norm_f, hn16, BL, 0);

    // lm_head: (BL x 256) x (256 x 20) -> d_out
    launch_gemm(hn16, D_MODEL, lm16, D_MODEL, out, 20, BL, 20, D_MODEL, nullptr, 0);
}
